// MultiheadAttention_54580444397760
// MI455X (gfx1250) — matrix operand order repaired and checked by reading, not compiled, not run on hardware
//
#include <hip/hip_runtime.h>
#include <hip/hip_bf16.h>

typedef __attribute__((ext_vector_type(16))) __bf16 v16bf;
typedef __attribute__((ext_vector_type(8)))  float  v8f;

union V16 {
    unsigned short h[16];
    v16bf v;
};

__device__ __forceinline__ unsigned short f2bf(float f) {
    union { float f; unsigned int u; } c;
    c.f = f;
    unsigned int u = c.u;
    unsigned int r = u + 0x7FFFu + ((u >> 16) & 1u);   // round-to-nearest-even
    return (unsigned short)(r >> 16);
}

// ---- WMMA fragment loaders (bf16, 16x16x32) ---------------------------------
// A-matrix 16x32 (MxK): lane L<16 -> M=L, K = {0..15}; lane>=16 -> M=L-16, K={16..31}
__device__ __forceinline__ v16bf load_a_tile(const unsigned short* p, int ld, int mBase, int kBase) {
    int lane = threadIdx.x & 31;
    int r = lane & 15, hi = lane >> 4;
    const unsigned short* row = p + (mBase + r) * ld + kBase;
    V16 x;
#pragma unroll
    for (int i = 0; i < 8; i++) x.h[i] = row[16 * hi + i];
#pragma unroll
    for (int i = 0; i < 8; i++) x.h[8 + i] = row[16 * hi + 8 + i];
    return x.v;
}

// B-matrix 32x16 (KxN) from storage organized [n][k] (e.g. W rows, K tiles)
__device__ __forceinline__ v16bf load_b_tile_nk(const unsigned short* p, int ld, int nBase, int kBase) {
    int lane = threadIdx.x & 31;
    int n = lane & 15, hi = lane >> 4;
    const unsigned short* row = p + (nBase + n) * ld + kBase + 16 * hi;
    V16 x;
#pragma unroll
    for (int i = 0; i < 16; i++) x.h[i] = row[i];
    return x.v;
}

// B-matrix 32x16 (KxN) from storage organized [k][n] (e.g. V tile)
__device__ __forceinline__ v16bf load_b_tile_kn(const unsigned short* p, int ld, int nBase, int kBase) {
    int lane = threadIdx.x & 31;
    int n = lane & 15, hi = lane >> 4;
    V16 x;
#pragma unroll
    for (int i = 0; i < 16; i++) x.h[i] = p[(kBase + 16 * hi + i) * ld + nBase + n];
    return x.v;
}

#define WMMA_BF16(a, b, c) \
    __builtin_amdgcn_wmma_f32_16x16x32_bf16(false, (a), false, (b), (short)0, (c), false, false)

// ---- fp32 -> bf16 convert ---------------------------------------------------
__global__ void mha_cvt_bf16(const float* __restrict__ in, unsigned short* __restrict__ out, int n) {
    int i = blockIdx.x * blockDim.x + threadIdx.x;
    if (i < n) out[i] = f2bf(in[i]);
}

// ---- bf16 WMMA GEMM: C[M,N] = (A[M,K] @ W[N,K]^T + bias) * scale ------------
#define BM 128
#define BN 128
#define BK 32
__global__ __launch_bounds__(256) void mha_gemm_bf16(
    const unsigned short* __restrict__ A,   // [M,K] row-major bf16
    const unsigned short* __restrict__ W,   // [N,K] row-major bf16 (nn.Linear weight)
    const float* __restrict__ bias,         // [N]
    float scale,
    unsigned short* __restrict__ outB,      // bf16 output (or null)
    float* __restrict__ outF,               // f32 output (or null)
    int M, int N, int K)
{
    __shared__ unsigned short As[BM][BK + 2];
    __shared__ unsigned short Bs[BN][BK + 2];

    const int tid = threadIdx.x;
    const int wid = tid >> 5, lane = tid & 31;
    const int m0 = blockIdx.y * BM, n0 = blockIdx.x * BN;
    const int wm = wid >> 2, wn = wid & 3;       // 2 x 4 wave grid, each wave: 64x32

    v8f c[4][2];
#pragma unroll
    for (int mt = 0; mt < 4; mt++)
#pragma unroll
        for (int nt = 0; nt < 2; nt++) c[mt][nt] = (v8f){0.f, 0.f, 0.f, 0.f, 0.f, 0.f, 0.f, 0.f};

    for (int k0 = 0; k0 < K; k0 += BK) {
        if (k0 + BK < K) {
            __builtin_prefetch(&A[(m0 + (tid >> 1)) * K + k0 + BK], 0, 1);
            __builtin_prefetch(&W[(n0 + (tid >> 1)) * K + k0 + BK], 0, 1);
        }
        for (int p = tid; p < BM * BK / 2; p += 256) {
            int r = p >> 4, kp = p & 15;
            *(unsigned int*)&As[r][kp * 2] = *(const unsigned int*)&A[(m0 + r) * K + k0 + kp * 2];
        }
        for (int p = tid; p < BN * BK / 2; p += 256) {
            int r = p >> 4, kp = p & 15;
            *(unsigned int*)&Bs[r][kp * 2] = *(const unsigned int*)&W[(n0 + r) * K + k0 + kp * 2];
        }
        __syncthreads();

        v16bf bt[2];
#pragma unroll
        for (int nt = 0; nt < 2; nt++)
            bt[nt] = load_b_tile_nk(&Bs[0][0], BK + 2, wn * 32 + nt * 16, 0);
#pragma unroll
        for (int mt = 0; mt < 4; mt++) {
            v16bf at = load_a_tile(&As[0][0], BK + 2, wm * 64 + mt * 16, 0);
#pragma unroll
            for (int nt = 0; nt < 2; nt++)
                c[mt][nt] = WMMA_BF16(at, bt[nt], c[mt][nt]);
        }
        __syncthreads();
    }

    // epilogue: hoist bias (invariant across the 8-row accumulator slice) and
    // the output-mode branch (uniform across the grid) out of the store loops
    const int hi = lane >> 4, nl = lane & 15;
    float bv[2];
#pragma unroll
    for (int nt = 0; nt < 2; nt++)
        bv[nt] = bias[n0 + wn * 32 + nt * 16 + nl];

    if (outB != nullptr) {
#pragma unroll
        for (int mt = 0; mt < 4; mt++) {
#pragma unroll
            for (int nt = 0; nt < 2; nt++) {
                int gn = n0 + wn * 32 + nt * 16 + nl;
#pragma unroll
                for (int i = 0; i < 8; i++) {
                    int gm = m0 + wm * 64 + mt * 16 + i + 8 * hi;
                    outB[gm * N + gn] = f2bf((c[mt][nt][i] + bv[nt]) * scale);
                }
            }
        }
    } else {
#pragma unroll
        for (int mt = 0; mt < 4; mt++) {
#pragma unroll
            for (int nt = 0; nt < 2; nt++) {
                int gn = n0 + wn * 32 + nt * 16 + nl;
#pragma unroll
                for (int i = 0; i < 8; i++) {
                    int gm = m0 + wm * 64 + mt * 16 + i + 8 * hi;
                    outF[gm * N + gn] = (c[mt][nt][i] + bv[nt]) * scale;
                }
            }
        }
    }
}

// ---- attention core ---------------------------------------------------------
// grid: (T/32, B*H), block: 128 threads (4 waves). Each block: 32 query rows, full S.
#define AT_T  1024
#define AT_B  8
#define AT_E  1024
#define AT_H  16
#define AT_HD 64
#define AT_S  1024
#define NEGI  (-1.0e30f)

__global__ __launch_bounds__(128) void mha_attn_kernel(
    const unsigned short* __restrict__ qs,   // bf16 [TOK, E] (already scaled)
    const unsigned short* __restrict__ ks,   // bf16 [TOK, E]
    const unsigned short* __restrict__ vs,   // bf16 [TOK, E]
    const unsigned char* __restrict__ kpad,  // bool [B, S]
    unsigned short* __restrict__ ctx,        // bf16 [TOK, E]
    float* __restrict__ wout)                // f32 [B*H, T, S] normalized weights
{
    const int bh = blockIdx.y;
    const int b = bh >> 4, h = bh & 15;
    const int t0 = blockIdx.x * 32;
    const int tid = threadIdx.x, wid = tid >> 5, lane = tid & 31;
    const int hi = lane >> 4, nl = lane & 15;
    const unsigned char* kpadb = kpad + b * AT_S;

    __shared__ unsigned short Qs[32][AT_HD + 4];        // 4352 B
    __shared__ unsigned short Kc[128][AT_HD + 4];       // 17408 B
    __shared__ __align__(16) unsigned char SV[17408];   // alias: Sc (f32 32x132) / Vc (bf16 128x68)
    __shared__ unsigned short Pc[32][136];              // 8704 B
    __shared__ float mrow[32], lrow[32];
    __shared__ float accS[32][AT_HD + 1];               // 8320 B

    float (*Sc)[132] = (float(*)[132])SV;
    unsigned short (*Vc)[AT_HD + 4] = (unsigned short(*)[AT_HD + 4])SV;

    // load Q tile (32 x 64) once
    for (int p = tid; p < 32 * AT_HD / 2; p += 128) {
        int r = p >> 5, dp = p & 31;
        *(unsigned int*)&Qs[r][dp * 2] =
            *(const unsigned int*)&qs[(((t0 + r) * AT_B + b) * AT_E) + h * AT_HD + dp * 2];
    }

    const int tmax = t0 + 31;
    const int nch = (tmax >> 7) + 1;          // chunks containing any causally valid column

    // ---------------- pass 1: online row max + row sum-of-exp ----------------
    float m_r = -3.0e38f, l_r = 0.0f;
    for (int cck = 0; cck < nch; cck++) {
        const int s0 = cck << 7;
        __syncthreads();
        for (int p = tid; p < 128 * AT_HD / 2; p += 128) {
            int r = p >> 5, dp = p & 31;
            *(unsigned int*)&Kc[r][dp * 2] =
                *(const unsigned int*)&ks[(((s0 + r) * AT_B + b) * AT_E) + h * AT_HD + dp * 2];
        }
        __syncthreads();
        // scores: wave 'wid' owns columns [32*wid, 32*wid+32)
#pragma unroll
        for (int mt = 0; mt < 2; mt++) {
            v16bf a0 = load_a_tile(&Qs[0][0], AT_HD + 4, mt * 16, 0);
            v16bf a1 = load_a_tile(&Qs[0][0], AT_HD + 4, mt * 16, 32);
#pragma unroll
            for (int nt = 0; nt < 2; nt++) {
                v16bf b0 = load_b_tile_nk(&Kc[0][0], AT_HD + 4, wid * 32 + nt * 16, 0);
                v16bf b1 = load_b_tile_nk(&Kc[0][0], AT_HD + 4, wid * 32 + nt * 16, 32);
                v8f sc = (v8f){0.f, 0.f, 0.f, 0.f, 0.f, 0.f, 0.f, 0.f};
                sc = WMMA_BF16(a0, b0, sc);
                sc = WMMA_BF16(a1, b1, sc);
#pragma unroll
                for (int i = 0; i < 8; i++)
                    Sc[mt * 16 + i + 8 * hi][wid * 32 + nt * 16 + nl] = sc[i];
            }
        }
        __syncthreads();
        if (tid < 32) {
            const int t = t0 + tid;
            int smax = t - s0; if (smax > 127) smax = 127;
            float mc = -3.0e38f;
            for (int s = 0; s <= smax; s++) {
                float v = kpadb[s0 + s] ? NEGI : Sc[tid][s];
                mc = fmaxf(mc, v);
            }
            float m_new = fmaxf(m_r, mc);
            l_r *= __expf(m_r - m_new);
            for (int s = 0; s <= smax; s++) {
                float v = kpadb[s0 + s] ? NEGI : Sc[tid][s];
                l_r += __expf(v - m_new);
            }
            m_r = m_new;
        }
    }
    if (tid < 32) {
        mrow[tid] = m_r;
        lrow[tid] = (l_r > 0.f) ? (1.0f / l_r) : 0.0f;
    }
    __syncthreads();

    // ---------------- pass 2: normalized weights out + P@V -------------------
    v8f acc[2][4];
#pragma unroll
    for (int mt = 0; mt < 2; mt++)
#pragma unroll
        for (int nt = 0; nt < 4; nt++) acc[mt][nt] = (v8f){0.f, 0.f, 0.f, 0.f, 0.f, 0.f, 0.f, 0.f};

    for (int cck = 0; cck < AT_S / 128; cck++) {
        const int s0 = cck << 7;
        if (s0 > tmax) {  // fully masked chunk: emit zero weights (uniform branch)
            for (int p = tid; p < 32 * 128; p += 128) {
                int r = p >> 7, s2 = p & 127;
                wout[((long)bh * AT_T + (t0 + r)) * AT_S + s0 + s2] = 0.0f;
            }
            continue;
        }
        __syncthreads();
        for (int p = tid; p < 128 * AT_HD / 2; p += 128) {
            int r = p >> 5, dp = p & 31;
            *(unsigned int*)&Kc[r][dp * 2] =
                *(const unsigned int*)&ks[(((s0 + r) * AT_B + b) * AT_E) + h * AT_HD + dp * 2];
        }
        __syncthreads();
#pragma unroll
        for (int mt = 0; mt < 2; mt++) {
            v16bf a0 = load_a_tile(&Qs[0][0], AT_HD + 4, mt * 16, 0);
            v16bf a1 = load_a_tile(&Qs[0][0], AT_HD + 4, mt * 16, 32);
#pragma unroll
            for (int nt = 0; nt < 2; nt++) {
                v16bf b0 = load_b_tile_nk(&Kc[0][0], AT_HD + 4, wid * 32 + nt * 16, 0);
                v16bf b1 = load_b_tile_nk(&Kc[0][0], AT_HD + 4, wid * 32 + nt * 16, 32);
                v8f sc = (v8f){0.f, 0.f, 0.f, 0.f, 0.f, 0.f, 0.f, 0.f};
                sc = WMMA_BF16(a0, b0, sc);
                sc = WMMA_BF16(a1, b1, sc);
#pragma unroll
                for (int i = 0; i < 8; i++)
                    Sc[mt * 16 + i + 8 * hi][wid * 32 + nt * 16 + nl] = sc[i];
            }
        }
        __syncthreads();
        // probabilities: p = exp(score - m) / l, matching NEG_INF = -1e30 masking
        for (int p = tid; p < 32 * 128; p += 128) {
            int r = p >> 7, s2 = p & 127;
            int t = t0 + r, sg = s0 + s2;
            float v = (sg <= t) ? (kpadb[sg] ? NEGI : Sc[r][s2]) : NEGI;
            float pr = __expf(v - mrow[r]) * lrow[r];
            wout[((long)bh * AT_T + t) * AT_S + sg] = pr;
            Pc[r][s2] = f2bf(pr);
        }
        __syncthreads();
        // V chunk (aliases score buffer; protected by the barrier above)
        for (int p = tid; p < 128 * AT_HD / 2; p += 128) {
            int r = p >> 5, dp = p & 31;
            *(unsigned int*)&Vc[r][dp * 2] =
                *(const unsigned int*)&vs[(((s0 + r) * AT_B + b) * AT_E) + h * AT_HD + dp * 2];
        }
        __syncthreads();
        // acc += P_chunk @ V_chunk ; wave 'wid' covers k-slice [32*wid, 32*wid+32)
#pragma unroll
        for (int mt = 0; mt < 2; mt++) {
            v16bf pa = load_a_tile(&Pc[0][0], 136, mt * 16, wid * 32);
#pragma unroll
            for (int nt = 0; nt < 4; nt++) {
                v16bf vb = load_b_tile_kn(&Vc[0][0], AT_HD + 4, nt * 16, wid * 32);
                acc[mt][nt] = WMMA_BF16(pa, vb, acc[mt][nt]);
            }
        }
    }

    // reduce partial accumulators across the 4 waves via LDS float atomics
    for (int p = tid; p < 32 * AT_HD; p += 128) accS[p >> 6][p & 63] = 0.0f;
    __syncthreads();
#pragma unroll
    for (int mt = 0; mt < 2; mt++)
#pragma unroll
        for (int nt = 0; nt < 4; nt++)
#pragma unroll
            for (int i = 0; i < 8; i++)
                atomicAdd(&accS[mt * 16 + i + 8 * hi][nt * 16 + nl], acc[mt][nt][i]);
    __syncthreads();
    for (int p = tid; p < 32 * AT_HD; p += 128) {
        int r = p >> 6, d = p & 63;
        ctx[(((t0 + r) * AT_B + b) * AT_E) + h * AT_HD + d] = f2bf(accS[r][d]);
    }
}

// ---- avg over heads: avg[b,t,s] = mean_h weights[b,h,t,s] -------------------
__global__ __launch_bounds__(256) void mha_avg_kernel(const float* __restrict__ w,
                                                      float* __restrict__ avg) {
    const int bt = blockIdx.x;            // b*T + t
    const int b = bt >> 10, t = bt & 1023;
    for (int s = threadIdx.x; s < 1024; s += 256) {
        float sum = 0.0f;
#pragma unroll
        for (int h = 0; h < 16; h++)
            sum += w[(((long)(b * 16 + h) * 1024) + t) * 1024 + s];
        avg[(long)bt * 1024 + s] = sum * (1.0f / 16.0f);
    }
}

extern "C" void kernel_launch(void* const* d_in, const int* in_sizes, int n_in,
                              void* d_out, int out_size, void* d_ws, size_t ws_size,
                              hipStream_t stream) {
    constexpr int T = 1024, B = 8, E = 1024, TOK = T * B;
    const float* query = (const float*)d_in[0];
    const float* key   = (const float*)d_in[1];
    const float* value = (const float*)d_in[2];
    const unsigned char* kpad = (const unsigned char*)d_in[3];
    // d_in[4] attn_mask: causal, computed analytically in-kernel
    const float* Wq = (const float*)d_in[5];
    const float* bq = (const float*)d_in[6];
    const float* Wk = (const float*)d_in[7];
    const float* bk = (const float*)d_in[8];
    const float* Wv = (const float*)d_in[9];
    const float* bv = (const float*)d_in[10];
    const float* Wo = (const float*)d_in[11];
    const float* bo = (const float*)d_in[12];

    unsigned short* qs   = (unsigned short*)d_ws;
    unsigned short* ksb  = qs  + (size_t)TOK * E;
    unsigned short* vsb  = ksb + (size_t)TOK * E;
    unsigned short* ctx  = vsb + (size_t)TOK * E;
    unsigned short* bufX = ctx + (size_t)TOK * E;
    unsigned short* wqb  = bufX + (size_t)TOK * E;
    unsigned short* wkb  = wqb + (size_t)E * E;
    unsigned short* wvb  = wkb + (size_t)E * E;
    unsigned short* wob  = wvb + (size_t)E * E;

    float* attn_out = (float*)d_out;                        // [T,B,E]
    float* avg_out  = attn_out + (size_t)TOK * E;           // [B,T,S]
    float* w_out    = avg_out + (size_t)B * T * T;          // [B,H,T,S]

    const int cvtBlk = 256;
    const int nW = E * E, nX = TOK * E;
    mha_cvt_bf16<<<(nW + cvtBlk - 1) / cvtBlk, cvtBlk, 0, stream>>>(Wq, wqb, nW);
    mha_cvt_bf16<<<(nW + cvtBlk - 1) / cvtBlk, cvtBlk, 0, stream>>>(Wk, wkb, nW);
    mha_cvt_bf16<<<(nW + cvtBlk - 1) / cvtBlk, cvtBlk, 0, stream>>>(Wv, wvb, nW);
    mha_cvt_bf16<<<(nW + cvtBlk - 1) / cvtBlk, cvtBlk, 0, stream>>>(Wo, wob, nW);

    dim3 gg(E / BN, TOK / BM);
    const float scaling = 0.125f;  // hd^-0.5, hd = 64

    mha_cvt_bf16<<<(nX + cvtBlk - 1) / cvtBlk, cvtBlk, 0, stream>>>(query, bufX, nX);
    mha_gemm_bf16<<<gg, 256, 0, stream>>>(bufX, wqb, bq, scaling, qs, nullptr, TOK, E, E);

    mha_cvt_bf16<<<(nX + cvtBlk - 1) / cvtBlk, cvtBlk, 0, stream>>>(key, bufX, nX);
    mha_gemm_bf16<<<gg, 256, 0, stream>>>(bufX, wkb, bk, 1.0f, ksb, nullptr, TOK, E, E);

    mha_cvt_bf16<<<(nX + cvtBlk - 1) / cvtBlk, cvtBlk, 0, stream>>>(value, bufX, nX);
    mha_gemm_bf16<<<gg, 256, 0, stream>>>(bufX, wvb, bv, 1.0f, vsb, nullptr, TOK, E, E);

    mha_attn_kernel<<<dim3(T / 32, B * 16), 128, 0, stream>>>(qs, ksb, vsb, kpad, ctx, w_out);

    mha_gemm_bf16<<<gg, 256, 0, stream>>>(ctx, wob, bo, 1.0f, nullptr, attn_out, TOK, E, E);

    mha_avg_kernel<<<B * T, 256, 0, stream>>>(w_out, avg_out);
}
